// SpaceSelfAttentionModality_37108517438309
// MI455X (gfx1250) — compile-verified, matrix operand order checked by reading, not run on hardware
//
#include <hip/hip_runtime.h>
#include <hip/hip_bf16.h>

// ---------------- types ----------------
typedef __attribute__((ext_vector_type(16))) _Float16 v16h;
typedef __attribute__((ext_vector_type(8)))  _Float16 v8h;
typedef __attribute__((ext_vector_type(4)))  _Float16 v4h;
typedef __attribute__((ext_vector_type(8)))  float    v8f;
typedef __attribute__((ext_vector_type(4)))  unsigned u32x4;
typedef __attribute__((ext_vector_type(4)))  int      i32x4;
typedef __attribute__((ext_vector_type(8)))  int      i32x8;

// Problem constants
#define BT   16        // B*T
#define SEQ  1024
#define DIM  1024
#define NH   16
#define HD   64
#define MTOT (BT*SEQ)  // 16384
#define NQKV (3*DIM)   // 3072

// ---------------- feature detection ----------------
#ifndef HAVE_TDM
#if defined(__has_builtin)
#if __has_builtin(__builtin_amdgcn_tensor_load_to_lds)
#define HAVE_TDM 1
#else
#define HAVE_TDM 0
#endif
#else
#define HAVE_TDM 0
#endif
#endif

#if defined(__has_builtin)
#if __has_builtin(__builtin_amdgcn_s_wait_tensorcnt)
#define HAVE_WAIT_TENSOR 1
#else
#define HAVE_WAIT_TENSOR 0
#endif
#else
#define HAVE_WAIT_TENSOR 0
#endif

template<int N>
static __device__ __forceinline__ void wait_tensor() {
#if HAVE_WAIT_TENSOR
    __builtin_amdgcn_s_wait_tensorcnt((short)N);
#else
    asm volatile("s_wait_tensorcnt %0" :: "i"(N));
#endif
}

#if HAVE_TDM
// Issue a TDM load of a 2D f16 tile: 32 contiguous halves x tile_d1 rows,
// row stride strideElems (f16 units), into LDS at byte offset lds_off.
// LDS destination gets 4 DWORDs (8 halves) of pad after every 16 DWORDs
// (32 halves) -> matches the LDT=40 padded tile layout.
static __device__ __forceinline__ void tdm_load_tile32(unsigned lds_off,
                                                       const _Float16* gptr,
                                                       unsigned tile_d1,
                                                       unsigned strideElems) {
    unsigned long long ga = (unsigned long long)gptr;
    u32x4 g0;
    g0[0] = 1u;                                           // count=1, user descriptor
    g0[1] = lds_off;                                      // lds_addr (bytes)
    g0[2] = (unsigned)ga;                                 // global_addr[31:0]
    g0[3] = (unsigned)((ga >> 32) & 0x01FFFFFFu) | (2u << 30);  // addr[56:32] | type=2
    i32x8 g1;
    // data_size=2B (code 1), pad_enable, pad_interval=16 DW (code 3), pad_amount=4 DW (code 3)
    g1[0] = (int)((1u << 16) | (1u << 20) | (3u << 22) | (3u << 25));
    g1[1] = (int)(32u << 16);          // tensor_dim0[15:0] = 32 (== tile_d0, no OOB)
    g1[2] = (int)(tile_d1 << 16);      // tensor_dim1[15:0] = tile_d1
    g1[3] = (int)(32u << 16);          // tile_dim0 = 32
    g1[4] = (int)tile_d1;              // tile_dim1
    g1[5] = (int)strideElems;          // tensor_dim0_stride[31:0]
    g1[6] = 0;
    g1[7] = 0;
    i32x4 z4 = {0, 0, 0, 0};
#if __clang_major__ >= 23
    i32x8 z8 = {0, 0, 0, 0, 0, 0, 0, 0};
    __builtin_amdgcn_tensor_load_to_lds(g0, g1, z4, z4, z8, 0);
#else
    __builtin_amdgcn_tensor_load_to_lds(g0, g1, z4, z4, 0);
#endif
}
#endif

static __device__ __forceinline__ v16h ld_frag(const _Float16* p0, const _Float16* p1) {
    v8h lo = *(const v8h*)p0;
    v8h hi = *(const v8h*)p1;
    return __builtin_shufflevector(lo, hi, 0,1,2,3,4,5,6,7,8,9,10,11,12,13,14,15);
}

static __device__ __forceinline__ v8f wmma16(v16h a, v16h b, v8f c) {
    // D = A(16x32 f16) * B(32x16 f16) + C(16x16 f32)
    return __builtin_amdgcn_wmma_f32_16x16x32_f16(false, a, false, b, (short)0, c, false, false);
}

static __device__ __forceinline__ int modid(int s) {
    // 64 latents(-1), 768 image(0), 32 action(1), 32 proprio(2), 128 register(3)
    return s < 64 ? -1 : (s < 832 ? 0 : (s < 864 ? 1 : (s < 896 ? 2 : 3)));
}

// ---------------- f32 -> f16 cast (x4 vectorized) ----------------
__global__ __launch_bounds__(256) void cast4_f32_f16(const float* __restrict__ in,
                                                     _Float16* __restrict__ out, int n4) {
    int i = blockIdx.x * 256 + threadIdx.x;
    if (i < n4) {
        float4 f = ((const float4*)in)[i];
        v4h o; o[0] = (_Float16)f.x; o[1] = (_Float16)f.y; o[2] = (_Float16)f.z; o[3] = (_Float16)f.w;
        ((v4h*)out)[i] = o;
    }
}

// ---------------- WMMA GEMM: out[M,N] = A[M,K] * W[N,K]^T + bias[N] ----------------
// 128x128 block tile, K-step 32, 8 waves: 4 (M) x 2 (N), each wave 32x64 (2x4 wmma tiles)
// Tiles staged by the Tensor Data Mover (double buffered) when available.
template<bool OUT_F16>
__global__ __launch_bounds__(256) void gemm_wmma(const _Float16* __restrict__ A,
                                                 const _Float16* __restrict__ W,
                                                 const float*    __restrict__ bias,
                                                 _Float16* __restrict__ outH,
                                                 float*    __restrict__ outF,
                                                 int M, int N, int K) {
    constexpr int LDT   = 40;            // 32 halves + 8 pad (80B rows, 16B aligned)
    constexpr int TILEH = 128 * LDT;     // halves per tile

    const int tid  = threadIdx.x;
    const int lane = tid & 31;
    const int wave = tid >> 5;
    const int wm   = wave & 3;           // 0..3 -> 32-row slab
    const int wn   = wave >> 2;          // 0..1 -> 64-col slab
    const int m0   = blockIdx.y * 128;
    const int n0   = blockIdx.x * 128;
    const int khalf = (lane >> 4) * 8;   // K interleave select per 16-bit A/B layout

    v8f acc[2][4];
    for (int i = 0; i < 2; i++)
        for (int j = 0; j < 4; j++)
            for (int r = 0; r < 8; r++) acc[i][j][r] = 0.0f;

#if HAVE_TDM
    // Buffer layout (LDS offset 0): A0 | W0 | A1 | W1
    __shared__ _Float16 tiles[4 * TILEH];
    const bool issuer = (tid < 32);

    if (issuer) {
        tdm_load_tile32(0u,                        A + (long)m0 * K, 128, (unsigned)K);
        tdm_load_tile32((unsigned)(TILEH * 2),     W + (long)n0 * K, 128, (unsigned)K);
    }
    int ib = 0;
    for (int kb = 0; kb < K; kb += 32, ib ^= 1) {
        if (issuer) {
            if (kb + 32 < K) {
                unsigned nb = (unsigned)(ib ^ 1);
                tdm_load_tile32((2 * nb)     * (unsigned)(TILEH * 2) / 2u * 2u, A + (long)m0 * K + kb + 32, 128, (unsigned)K);
                tdm_load_tile32((2 * nb + 1) * (unsigned)(TILEH * 2),           W + (long)n0 * K + kb + 32, 128, (unsigned)K);
                wait_tensor<2>();     // current buffer's 2 loads complete (in-order)
            } else {
                wait_tensor<0>();
            }
        }
        __syncthreads();              // DMA'd tiles visible to all waves
        const _Float16* Acur = tiles + (2 * ib) * TILEH;
        const _Float16* Wcur = tiles + (2 * ib + 1) * TILEH;

        v16h af[2], bf[4];
        #pragma unroll
        for (int i = 0; i < 2; i++) {
            const _Float16* p = &Acur[(wm * 32 + i * 16 + (lane & 15)) * LDT];
            af[i] = ld_frag(p + khalf, p + 16 + khalf);
        }
        #pragma unroll
        for (int j = 0; j < 4; j++) {
            const _Float16* p = &Wcur[(wn * 64 + j * 16 + (lane & 15)) * LDT];
            bf[j] = ld_frag(p + khalf, p + 16 + khalf);
        }
        #pragma unroll
        for (int i = 0; i < 2; i++)
            #pragma unroll
            for (int j = 0; j < 4; j++)
                acc[i][j] = wmma16(af[i], bf[j], acc[i][j]);
        __syncthreads();              // reads done before next DMA overwrites
    }
#else
    __shared__ _Float16 As[TILEH];
    __shared__ _Float16 Ws[TILEH];
    for (int kb = 0; kb < K; kb += 32) {
        __syncthreads();
        #pragma unroll
        for (int i = 0; i < 2; i++) {
            int v = tid + i * 256;
            int r = v >> 2, c = (v & 3) * 8;
            *(v8h*)&As[r * LDT + c] = *(const v8h*)&A[(long)(m0 + r) * K + kb + c];
            *(v8h*)&Ws[r * LDT + c] = *(const v8h*)&W[(long)(n0 + r) * K + kb + c];
        }
        __syncthreads();
        v16h af[2], bf[4];
        #pragma unroll
        for (int i = 0; i < 2; i++) {
            const _Float16* p = &As[(wm * 32 + i * 16 + (lane & 15)) * LDT];
            af[i] = ld_frag(p + khalf, p + 16 + khalf);
        }
        #pragma unroll
        for (int j = 0; j < 4; j++) {
            const _Float16* p = &Ws[(wn * 64 + j * 16 + (lane & 15)) * LDT];
            bf[j] = ld_frag(p + khalf, p + 16 + khalf);
        }
        #pragma unroll
        for (int i = 0; i < 2; i++)
            #pragma unroll
            for (int j = 0; j < 4; j++)
                acc[i][j] = wmma16(af[i], bf[j], acc[i][j]);
    }
#endif

    // C/D layout: lane -> n (lane&15), (lane>>4)*8 + r -> m
    #pragma unroll
    for (int i = 0; i < 2; i++)
        #pragma unroll
        for (int j = 0; j < 4; j++) {
            int n  = n0 + wn * 64 + j * 16 + (lane & 15);
            int mb = m0 + wm * 32 + i * 16 + (lane >> 4) * 8;
            float b = bias[n];
            #pragma unroll
            for (int r = 0; r < 8; r++) {
                float val = acc[i][j][r] + b;
                long idx = (long)(mb + r) * N + n;
                if (OUT_F16) outH[idx] = (_Float16)val;
                else         outF[idx] = val;
            }
        }
}

// ---------------- Flash attention (one block = 128 queries x 1 head x 1 bt) ----------------
// qkvh: [MTOT, 3*DIM] f16 (q | k | v). attnh: [MTOT, DIM] f16 (head-concat output).
__global__ __launch_bounds__(256) void attn_wmma(const _Float16* __restrict__ qkvh,
                                                 _Float16* __restrict__ attnh) {
    constexpr int LQ = 72;    // Q/K row stride (64 + 8 pad halves, 144B = 16B aligned)
    constexpr int LP = 136;   // Vt / P row stride (128 + 8 pad, 272B = 16B aligned)
    extern __shared__ _Float16 smem[];
    _Float16* Qs = smem;                       // 128 x 72
    _Float16* Ks = Qs + 128 * LQ;              // 128 x 72
    _Float16* Vt = Ks + 128 * LQ;              // 64  x 136 (transposed V: [dim][key])
    _Float16* Ps = Vt + 64 * LP;               // 8 waves x 16 x 136

    const int tid  = threadIdx.x;
    const int lane = tid & 31;
    const int wave = tid >> 5;
    const int qb   = blockIdx.x;               // 0..7   (query block)
    const int h    = blockIdx.y;               // 0..15  (head)
    const int bt   = blockIdx.z;               // 0..15
    const long rowBase = (long)bt * SEQ;
    const int qcol = h * HD;
    const int kcol = DIM + h * HD;
    const int vcol = 2 * DIM + h * HD;
    const int khalf = (lane >> 4) * 8;

    // stage Q block (128 x 64 halves)
    for (int i = tid; i < 128 * 8; i += 256) {
        int r = i >> 3, c = (i & 7) * 8;
        *(v8h*)&Qs[r * LQ + c] = *(const v8h*)&qkvh[(rowBase + qb * 128 + r) * NQKV + qcol + c];
    }
    __syncthreads();

    v16h qf[2];
    #pragma unroll
    for (int k2 = 0; k2 < 2; k2++) {
        const _Float16* p = &Qs[(wave * 16 + (lane & 15)) * LQ + k2 * 32];
        qf[k2] = ld_frag(p + khalf, p + 16 + khalf);
    }

    float mrow[8], lrow[8];
    v8f accO[4];
    #pragma unroll
    for (int r = 0; r < 8; r++) { mrow[r] = -1e30f; lrow[r] = 0.0f; }
    #pragma unroll
    for (int j = 0; j < 4; j++)
        for (int r = 0; r < 8; r++) accO[j][r] = 0.0f;

    _Float16* Pw = Ps + wave * 16 * LP;
    const int qloc = qb * 128 + wave * 16 + (lane >> 4) * 8;   // + r = global query index in S

    for (int kb = 0; kb < SEQ; kb += 128) {
        __syncthreads();   // previous iteration's reads of Ks/Vt done
        // stage K (row-major) and V (transposed); prefetch next block into L2
        for (int i = tid; i < 128 * 8; i += 256) {
            int r = i >> 3, c = (i & 7) * 8;
            *(v8h*)&Ks[r * LQ + c] = *(const v8h*)&qkvh[(rowBase + kb + r) * NQKV + kcol + c];
            v8h v = *(const v8h*)&qkvh[(rowBase + kb + r) * NQKV + vcol + c];
            #pragma unroll
            for (int e = 0; e < 8; e++) Vt[(c + e) * LP + r] = v[e];
            if (kb + 128 < SEQ) {
                __builtin_prefetch(&qkvh[(rowBase + kb + 128 + r) * NQKV + kcol + c], 0, 1);
                __builtin_prefetch(&qkvh[(rowBase + kb + 128 + r) * NQKV + vcol + c], 0, 1);
            }
        }
        __syncthreads();

        // S = Q * K^T : 16 (rows/wave) x 128 (keys) in 8 wmma tiles, 2 K-steps
        v8f accS[8];
        #pragma unroll
        for (int j = 0; j < 8; j++)
            for (int r = 0; r < 8; r++) accS[j][r] = 0.0f;
        #pragma unroll
        for (int k2 = 0; k2 < 2; k2++) {
            #pragma unroll
            for (int j = 0; j < 8; j++) {
                const _Float16* p = &Ks[(j * 16 + (lane & 15)) * LQ + k2 * 32];
                v16h bf = ld_frag(p + khalf, p + 16 + khalf);
                accS[j] = wmma16(qf[k2], bf, accS[j]);
            }
        }

        // mask + scale (1/sqrt(64)) + online softmax
        float bmax[8];
        #pragma unroll
        for (int r = 0; r < 8; r++) bmax[r] = -1e30f;
        #pragma unroll
        for (int j = 0; j < 8; j++) {
            int kg = kb + j * 16 + (lane & 15);
            int mk = modid(kg);
            #pragma unroll
            for (int r = 0; r < 8; r++) {
                int qg = qloc + r;
                bool allow = (qg < 64) ? (kg < 64) : ((modid(qg) == mk) || (kg < 64));
                float x = allow ? accS[j][r] * 0.125f : -1e30f;
                accS[j][r] = x;
                bmax[r] = fmaxf(bmax[r], x);
            }
        }
        // row lives inside one 16-lane half: reduce across lanes 0..15 / 16..31
        #pragma unroll
        for (int r = 0; r < 8; r++)
            for (int off = 1; off < 16; off <<= 1)
                bmax[r] = fmaxf(bmax[r], __shfl_xor(bmax[r], off, 16));

        float corr[8], rsum[8];
        #pragma unroll
        for (int r = 0; r < 8; r++) {
            float nm = fmaxf(mrow[r], bmax[r]);
            corr[r] = __expf(mrow[r] - nm);
            mrow[r] = nm;
            rsum[r] = 0.0f;
        }
        #pragma unroll
        for (int j = 0; j < 8; j++) {
            #pragma unroll
            for (int r = 0; r < 8; r++) {
                float p = __expf(accS[j][r] - mrow[r]);
                rsum[r] += p;
                Pw[((lane >> 4) * 8 + r) * LP + j * 16 + (lane & 15)] = (_Float16)p;
            }
        }
        #pragma unroll
        for (int r = 0; r < 8; r++) {
            for (int off = 1; off < 16; off <<= 1)
                rsum[r] += __shfl_xor(rsum[r], off, 16);
            lrow[r] = lrow[r] * corr[r] + rsum[r];
        }
        #pragma unroll
        for (int j = 0; j < 4; j++)
            for (int r = 0; r < 8; r++) accO[j][r] *= corr[r];
        __syncthreads();   // P visible (and Vt stable)

        // O += P(16x128) * V(128x64): 4 K-steps x 4 n-tiles
        #pragma unroll
        for (int kc = 0; kc < 4; kc++) {
            const _Float16* pp = &Pw[(lane & 15) * LP + kc * 32];
            v16h pf = ld_frag(pp + khalf, pp + 16 + khalf);
            #pragma unroll
            for (int j = 0; j < 4; j++) {
                const _Float16* pv = &Vt[(j * 16 + (lane & 15)) * LP + kc * 32];
                v16h vf = ld_frag(pv + khalf, pv + 16 + khalf);
                accO[j] = wmma16(pf, vf, accO[j]);
            }
        }
    }

    // epilogue: normalize and write head slice as f16
    #pragma unroll
    for (int j = 0; j < 4; j++) {
        int col = h * HD + j * 16 + (lane & 15);
        #pragma unroll
        for (int r = 0; r < 8; r++) {
            long row = rowBase + qb * 128 + wave * 16 + (lane >> 4) * 8 + r;
            attnh[row * DIM + col] = (_Float16)(accO[j][r] / lrow[r]);
        }
    }
}

// ---------------- launch ----------------
extern "C" void kernel_launch(void* const* d_in, const int* in_sizes, int n_in,
                              void* d_out, int out_size, void* d_ws, size_t ws_size,
                              hipStream_t stream) {
    const float* x      = (const float*)d_in[0];   // [16,1024,1024] (BT,S,D)
    const float* w_qkv  = (const float*)d_in[1];   // [3072,1024]
    const float* b_qkv  = (const float*)d_in[2];   // [3072]
    const float* w_out  = (const float*)d_in[3];   // [1024,1024]
    const float* b_out  = (const float*)d_in[4];   // [1024]
    // d_in[5] = mask -- recomputed analytically on device

    char* w = (char*)d_ws;
    _Float16* xh    = (_Float16*)w; w += (size_t)MTOT * DIM  * 2;   // 32 MB
    _Float16* wqkvh = (_Float16*)w; w += (size_t)NQKV * DIM  * 2;   // 6 MB
    _Float16* wouth = (_Float16*)w; w += (size_t)DIM  * DIM  * 2;   // 2 MB
    _Float16* qkvh  = (_Float16*)w; w += (size_t)MTOT * NQKV * 2;   // 96 MB
    _Float16* attnh = (_Float16*)w;                                 // 32 MB

    // 1) casts
    {
        int n4 = MTOT * DIM / 4;
        cast4_f32_f16<<<n4 / 256, 256, 0, stream>>>(x, xh, n4);
        n4 = NQKV * DIM / 4;
        cast4_f32_f16<<<n4 / 256, 256, 0, stream>>>(w_qkv, wqkvh, n4);
        n4 = DIM * DIM / 4;
        cast4_f32_f16<<<n4 / 256, 256, 0, stream>>>(w_out, wouth, n4);
    }

    // 2) QKV projection: [16384,3072] = xh * w_qkv^T + b_qkv  (f16 out)
    gemm_wmma<true><<<dim3(NQKV / 128, MTOT / 128), 256, 0, stream>>>(
        xh, wqkvh, b_qkv, qkvh, nullptr, MTOT, NQKV, DIM);

    // 3) masked flash attention per (query-block, head, bt)
    {
        size_t lds = (size_t)(128 * 72 + 128 * 72 + 64 * 136 + 8 * 16 * 136) * sizeof(_Float16);
        attn_wmma<<<dim3(SEQ / 128, NH, BT), 256, lds, stream>>>(qkvh, attnh);
    }

    // 4) output projection: d_out = attnh * w_out^T + b_out (f32 out)
    gemm_wmma<false><<<dim3(DIM / 128, MTOT / 128), 256, 0, stream>>>(
        attnh, wouth, b_out, nullptr, (float*)d_out, MTOT, DIM, DIM);
}
